// TransformerBlock_8899172237943
// MI455X (gfx1250) — matrix operand order repaired and checked by reading, not compiled, not run on hardware
//
#include <hip/hip_runtime.h>
#include <hip/hip_bf16.h>
#include <math.h>

typedef __attribute__((ext_vector_type(16))) __bf16 bf16x16;
typedef __attribute__((ext_vector_type(8)))  __bf16 bf16x8;
typedef __attribute__((ext_vector_type(8)))  float  f32x8;
typedef int v4i __attribute__((vector_size(16)));   // matches builtin param type

// ---- CDNA5 async global->LDS copy (ASYNCcnt) with sync fallback ------------
#if defined(__HIP_DEVICE_COMPILE__) && __has_builtin(__builtin_amdgcn_global_load_async_to_lds_b128)
#define HAVE_ASYNC 1
#else
#define HAVE_ASYNC 0
#endif

#if HAVE_ASYNC
#define WAIT_ASYNC() asm volatile("s_wait_asynccnt 0x0" ::: "memory")
#else
#define WAIT_ASYNC()
#endif

__device__ __forceinline__ void copy16(const __bf16* g, __bf16* l) {
#if HAVE_ASYNC
    typedef __attribute__((address_space(1))) v4i* gp_t;
    typedef __attribute__((address_space(3))) v4i* lp_t;
    __builtin_amdgcn_global_load_async_to_lds_b128(
        (gp_t)(unsigned long long)(size_t)g,
        (lp_t)(unsigned int)(size_t)l,
        0, 0);
#else
    *(bf16x8*)l = *(const bf16x8*)g;
#endif
}

// Assemble a 16-element fragment from two contiguous 16B chunks.
// Valid for both A (chunks at +8h and +16+8h) and B (chunks at +8h and +16+8h)
// per the CDNA5 ISA 16-bit VGPR layouts.
__device__ __forceinline__ bf16x16 mk16(bf16x8 lo, bf16x8 hi) {
    bf16x16 v;
#pragma unroll
    for (int i = 0; i < 8; ++i) { v[i] = lo[i]; v[i + 8] = hi[i]; }
    return v;
}
__device__ __forceinline__ bf16x16 frag_a(const __bf16* p, int hh) {
    return mk16(*(const bf16x8*)(p + 8 * hh), *(const bf16x8*)(p + 16 + 8 * hh));
}
__device__ __forceinline__ bf16x16 frag_b(const __bf16* p, int hh) {
    return mk16(*(const bf16x8*)(p + 8 * hh), *(const bf16x8*)(p + 16 + 8 * hh));
}

__device__ __forceinline__ float gelu_f(float x) {
    float x3 = x * x * x;
    return 0.5f * x * (1.0f + tanhf(0.7978845608028654f * (x + 0.044715f * x3)));
}

// ---------------- weight convert+transpose: Wt[n][k] = bf16(W[k][n]) --------
__global__ __launch_bounds__(256) void wtrans_kernel(const float* __restrict__ in,
                                                     __bf16* __restrict__ out,
                                                     int K, int N) {
    __shared__ float t[32][33];
    int nb = blockIdx.x * 32, kb = blockIdx.y * 32;
    int tx = threadIdx.x & 31, ty = threadIdx.x >> 5;   // ty: 0..7
#pragma unroll
    for (int i = 0; i < 32; i += 8)
        t[ty + i][tx] = in[(size_t)(kb + ty + i) * N + nb + tx];
    __syncthreads();
#pragma unroll
    for (int i = 0; i < 32; i += 8)
        out[(size_t)(nb + ty + i) * K + kb + tx] = (__bf16)t[tx][ty + i];
}

// ---------------- V transpose: vt[bh][d][s] = qkv_v[b][s][h*64+d] (bf16) ----
__global__ __launch_bounds__(256) void vtrans_kernel(const __bf16* __restrict__ qkv,
                                                     __bf16* __restrict__ vt) {
    __shared__ __bf16 t[32][33];
    int bh = blockIdx.z;
    int b = bh / 12, head = bh % 12;
    int s0 = blockIdx.x * 32, d0 = blockIdx.y * 32;
    int tx = threadIdx.x & 31, ty = threadIdx.x >> 5;
    const __bf16* src = qkv + (size_t)b * 2048 * 2304 + 1536 + head * 64;
#pragma unroll
    for (int i = 0; i < 32; i += 8)
        t[ty + i][tx] = src[(size_t)(s0 + ty + i) * 2304 + d0 + tx];
    __syncthreads();
    __bf16* dst = vt + ((size_t)bh * 64 + d0) * 2048 + s0;
#pragma unroll
    for (int i = 0; i < 32; i += 8)
        dst[(size_t)(ty + i) * 2048 + tx] = t[tx][ty + i];
}

// ---------------- LayerNorm -> bf16 -----------------------------------------
__global__ __launch_bounds__(256) void ln_kernel(const float* __restrict__ x,
                                                 const float* __restrict__ w,
                                                 const float* __restrict__ b,
                                                 __bf16* __restrict__ out) {
    int row = blockIdx.x;
    int tid = threadIdx.x;
    const float* xr = x + (size_t)row * 768;
    float v0 = xr[tid], v1 = xr[tid + 256], v2 = xr[tid + 512];

    __shared__ float red[256];
    red[tid] = v0 + v1 + v2;
    __syncthreads();
#pragma unroll
    for (int off = 128; off > 0; off >>= 1) {
        if (tid < off) red[tid] += red[tid + off];
        __syncthreads();
    }
    float mean = red[0] * (1.0f / 768.0f);
    __syncthreads();
    red[tid] = v0 * v0 + v1 * v1 + v2 * v2;
    __syncthreads();
#pragma unroll
    for (int off = 128; off > 0; off >>= 1) {
        if (tid < off) red[tid] += red[tid + off];
        __syncthreads();
    }
    float var = red[0] * (1.0f / 768.0f) - mean * mean;
    float inv = rsqrtf(var + 1e-5f);

    __bf16* orow = out + (size_t)row * 768;
    orow[tid]       = (__bf16)((v0 - mean) * inv * w[tid]       + b[tid]);
    orow[tid + 256] = (__bf16)((v1 - mean) * inv * w[tid + 256] + b[tid + 256]);
    orow[tid + 512] = (__bf16)((v2 - mean) * inv * w[tid + 512] + b[tid + 512]);
}

// ---------------- bf16 WMMA GEMM, async double-buffered tiles ----------------
// A:[M,K] bf16 row-major. Wt:[N,K] bf16 row-major (pre-transposed weights).
// C = act(A@W + bias [+ res]); OUTBF selects bf16 or fp32 output.
#define BM 64
#define BN 128
#define BK 32
#define LDP (BK + 8)    // padded LDS row: 80B, keeps 16B chunk alignment

__device__ __forceinline__ void tile_issue(const __bf16* __restrict__ A,
                                           const __bf16* __restrict__ Wt,
                                           __bf16 (*as)[LDP], __bf16 (*ws)[LDP],
                                           int tid, int mBase, int nBase, int K, int k0) {
    // A tile: 64 rows x 32 k = 256 16B chunks, one per thread
    int r = tid >> 2, c = (tid & 3) * 8;
    copy16(A + (size_t)(mBase + r) * K + k0 + c, &as[r][c]);
    // Wt tile: 128 rows x 32 k = 512 chunks, two per thread
#pragma unroll
    for (int t = 0; t < 2; ++t) {
        int idx = tid + t * 256;
        int rw = idx >> 2, cw = (idx & 3) * 8;
        copy16(Wt + (size_t)(nBase + rw) * K + k0 + cw, &ws[rw][cw]);
    }
}

template <bool RES, bool ACT, bool OUTBF>
__global__ __launch_bounds__(256) void gemm_kernel(const __bf16* __restrict__ A,
                                                   const __bf16* __restrict__ Wt,
                                                   const float* __restrict__ bias,
                                                   const float* __restrict__ res,
                                                   void* __restrict__ Cv,
                                                   int M, int N, int K) {
    __shared__ __bf16 As[2][BM][LDP];
    __shared__ __bf16 Ws[2][BN][LDP];

    int tid  = threadIdx.x;
    int lane = tid & 31;
    int wave = tid >> 5;
    int hh   = lane >> 4;
    int ln16 = lane & 15;
    int wm   = wave >> 2;   // 0..1
    int wn   = wave & 3;    // 0..3
    int mBase = blockIdx.y * BM;
    int nBase = blockIdx.x * BN;

    f32x8 zero = {0.f, 0.f, 0.f, 0.f, 0.f, 0.f, 0.f, 0.f};
    f32x8 acc[2][2];
    acc[0][0] = zero; acc[0][1] = zero; acc[1][0] = zero; acc[1][1] = zero;

    int nk = K / BK;
    tile_issue(A, Wt, As[0], Ws[0], tid, mBase, nBase, K, 0);

    for (int it = 0; it < nk; ++it) {
        int cur = it & 1;
        WAIT_ASYNC();
        __syncthreads();
        if (it + 1 < nk)
            tile_issue(A, Wt, As[cur ^ 1], Ws[cur ^ 1], tid, mBase, nBase, K, (it + 1) * BK);

        bf16x16 afrag[2], bfrag[2];
#pragma unroll
        for (int i = 0; i < 2; ++i)
            afrag[i] = frag_a(&As[cur][wm * 32 + i * 16 + ln16][0], hh);
#pragma unroll
        for (int j = 0; j < 2; ++j)
            bfrag[j] = frag_b(&Ws[cur][wn * 32 + j * 16 + ln16][0], hh);

#pragma unroll
        for (int i = 0; i < 2; ++i)
#pragma unroll
            for (int j = 0; j < 2; ++j)
                acc[i][j] = __builtin_amdgcn_wmma_f32_16x16x32_bf16(
                    false, afrag[i], false, bfrag[j], (short)0, acc[i][j], false, false);
    }

    // epilogue: C/D element r -> row r+8*hh, col = lane&15
#pragma unroll
    for (int i = 0; i < 2; ++i) {
#pragma unroll
        for (int j = 0; j < 2; ++j) {
            int col = nBase + wn * 32 + j * 16 + ln16;
            float bval = bias[col];
#pragma unroll
            for (int r = 0; r < 8; ++r) {
                int row = mBase + wm * 32 + i * 16 + r + 8 * hh;
                float v = acc[i][j][r] + bval;
                if (RES) v += res[(size_t)row * N + col];
                if (ACT) v = gelu_f(v);
                if (OUTBF) ((__bf16*)Cv)[(size_t)row * N + col] = (__bf16)v;
                else       ((float*)Cv)[(size_t)row * N + col] = v;
            }
        }
    }
}

// ---------------- causal flash attention (bf16 in/out, fp32 softmax) --------
// qkv: [B,S,2304] bf16. vt: [B*H,64,S] bf16. out: [B,S,768] bf16.
__global__ __launch_bounds__(128) void attn_kernel(const __bf16* __restrict__ qkv,
                                                   const __bf16* __restrict__ vt,
                                                   __bf16* __restrict__ out) {
    const int S = 2048, D3 = 2304, Dm = 768;
    int bh   = blockIdx.y;
    int b    = bh / 12;
    int head = bh % 12;
    int wave = threadIdx.x >> 5;
    int lane = threadIdx.x & 31;
    int hh   = lane >> 4;
    int ln16 = lane & 15;
    int q0   = blockIdx.x * 64 + wave * 16;

    const __bf16* base = qkv + (size_t)b * S * D3;
    const __bf16* qb   = base + head * 64;
    const __bf16* kbp  = base + 768 + head * 64;
    const __bf16* vtb  = vt + (size_t)bh * 64 * S;

    __shared__ __bf16 Plds[4][16][40];   // per-wave P tile, 80B rows (16B aligned)

    // Q A-fragments (K-dim 0..31 and 32..63): contiguous b128 chunks
    bf16x16 qf[2];
    {
        const __bf16* qrow = qb + (size_t)(q0 + ln16) * D3;
#pragma unroll
        for (int t = 0; t < 2; ++t)
            qf[t] = mk16(*(const bf16x8*)(qrow + 32 * t + 8 * hh),
                         *(const bf16x8*)(qrow + 32 * t + 16 + 8 * hh));
    }

    f32x8 zero = {0.f, 0.f, 0.f, 0.f, 0.f, 0.f, 0.f, 0.f};
    float mrow[8], lrow[8];
#pragma unroll
    for (int r = 0; r < 8; ++r) { mrow[r] = -3.0e38f; lrow[r] = 0.f; }
    f32x8 o[4];
#pragma unroll
    for (int t = 0; t < 4; ++t) o[t] = zero;

    for (int kb0 = 0; kb0 < q0 + 16; kb0 += 32) {
        // ---- S = Q @ K^T : two 16x16 tiles (B frag = contiguous K rows) ----
        f32x8 sA[2];
        sA[0] = zero; sA[1] = zero;
#pragma unroll
        for (int j = 0; j < 2; ++j) {
            const __bf16* krow = kbp + (size_t)(kb0 + j * 16 + ln16) * D3;
#pragma unroll
            for (int t = 0; t < 2; ++t) {
                bf16x16 kf = mk16(*(const bf16x8*)(krow + 32 * t + 8 * hh),
                                  *(const bf16x8*)(krow + 32 * t + 16 + 8 * hh));
                sA[j] = __builtin_amdgcn_wmma_f32_16x16x32_bf16(
                    false, qf[t], false, kf, (short)0, sA[j], false, false);
            }
        }
        // ---- scale + causal mask ----
#pragma unroll
        for (int j = 0; j < 2; ++j) {
#pragma unroll
            for (int r = 0; r < 8; ++r) {
                int qi = q0 + r + 8 * hh;
                int ki = kb0 + j * 16 + ln16;
                float s = sA[j][r] * 0.125f;   // 1/sqrt(64)
                sA[j][r] = (ki <= qi) ? s : -3.0e38f;
            }
        }
        // ---- online softmax; each row lives in one 16-lane half-wave ----
        float alpha[8];
#pragma unroll
        for (int r = 0; r < 8; ++r) {
            float mx = fmaxf(sA[0][r], sA[1][r]);
#pragma unroll
            for (int off = 8; off > 0; off >>= 1)
                mx = fmaxf(mx, __shfl_xor(mx, off, 32));
            float mnew = fmaxf(mrow[r], mx);
            float a  = __expf(mrow[r] - mnew);
            float p0 = __expf(sA[0][r] - mnew);
            float p1 = __expf(sA[1][r] - mnew);
            sA[0][r] = p0; sA[1][r] = p1;
            float sum = p0 + p1;
#pragma unroll
            for (int off = 8; off > 0; off >>= 1)
                sum += __shfl_xor(sum, off, 32);
            lrow[r] = lrow[r] * a + sum;
            mrow[r] = mnew;
            alpha[r] = a;
        }
#pragma unroll
        for (int t = 0; t < 4; ++t)
#pragma unroll
            for (int r = 0; r < 8; ++r) o[t][r] *= alpha[r];

        // ---- P: D-layout -> A-layout via per-wave LDS transpose ----
#pragma unroll
        for (int j = 0; j < 2; ++j)
#pragma unroll
            for (int r = 0; r < 8; ++r)
                Plds[wave][r + 8 * hh][j * 16 + ln16] = (__bf16)sA[j][r];
        asm volatile("s_wait_dscnt 0x0" ::: "memory");   // in-wave LDS visibility
        const __bf16* prow = &Plds[wave][ln16][0];
        bf16x16 pf = mk16(*(const bf16x8*)(prow + 8 * hh),
                          *(const bf16x8*)(prow + 16 + 8 * hh));

        // ---- O += P @ V : V^T rows give contiguous b128 B-fragments ----
#pragma unroll
        for (int t = 0; t < 4; ++t) {
            const __bf16* vrow = vtb + (size_t)(t * 16 + ln16) * S + kb0;
            bf16x16 vf = mk16(*(const bf16x8*)(vrow + 8 * hh),
                              *(const bf16x8*)(vrow + 16 + 8 * hh));
            o[t] = __builtin_amdgcn_wmma_f32_16x16x32_bf16(
                false, pf, false, vf, (short)0, o[t], false, false);
        }
    }

    // ---- write normalized bf16 output ----
#pragma unroll
    for (int t = 0; t < 4; ++t) {
#pragma unroll
        for (int r = 0; r < 8; ++r) {
            int row = q0 + r + 8 * hh;
            out[(size_t)(b * S + row) * Dm + head * 64 + t * 16 + ln16] =
                (__bf16)(o[t][r] * (1.0f / lrow[r]));
        }
    }
}

// ---------------- Driver ----------------------------------------------------
extern "C" void kernel_launch(void* const* d_in, const int* in_sizes, int n_in,
                              void* d_out, int out_size, void* d_ws, size_t ws_size,
                              hipStream_t stream) {
    const int Bb = 4, S = 2048, Dd = 768;
    const size_t BS = (size_t)Bb * S;   // 8192 rows

    const float* x     = (const float*)d_in[0];
    const float* ln1_w = (const float*)d_in[1];
    const float* ln1_b = (const float*)d_in[2];
    const float* w_attn = (const float*)d_in[3];
    const float* b_attn = (const float*)d_in[4];
    const float* w_proj = (const float*)d_in[5];
    const float* b_proj = (const float*)d_in[6];
    const float* ln2_w = (const float*)d_in[7];
    const float* ln2_b = (const float*)d_in[8];
    const float* w_fc  = (const float*)d_in[9];
    const float* b_fc  = (const float*)d_in[10];
    const float* w_fcp = (const float*)d_in[11];
    const float* b_fcp = (const float*)d_in[12];

    char* p = (char*)d_ws;
    __bf16* h       = (__bf16*)p;  p += BS * 768 * 2;      // LN out (bf16)
    float*  x1      = (float*)p;   p += BS * 768 * 4;      // post-attn residual
    __bf16* qkv     = (__bf16*)p;  p += BS * 2304 * 2;     // qkv (bf16)
    __bf16* attnout = (__bf16*)p;  p += BS * 768 * 2;      // attention out
    __bf16* fc      = qkv;                                  // [BS,3072] aliases qkv+attnout
    __bf16* vt      = (__bf16*)p;  p += BS * 768 * 2;      // V transposed
    __bf16* wt_attn = (__bf16*)p;  p += (size_t)768 * 2304 * 2;
    __bf16* wt_proj = (__bf16*)p;  p += (size_t)768 * 768 * 2;
    __bf16* wt_fc   = (__bf16*)p;  p += (size_t)768 * 3072 * 2;
    __bf16* wt_fcp  = (__bf16*)p;  p += (size_t)3072 * 768 * 2;

    // 0) weights -> bf16, transposed to [N,K]
    wtrans_kernel<<<dim3(2304 / 32, 768 / 32), 256, 0, stream>>>(w_attn, wt_attn, 768, 2304);
    wtrans_kernel<<<dim3(768 / 32, 768 / 32), 256, 0, stream>>>(w_proj, wt_proj, 768, 768);
    wtrans_kernel<<<dim3(3072 / 32, 768 / 32), 256, 0, stream>>>(w_fc, wt_fc, 768, 3072);
    wtrans_kernel<<<dim3(768 / 32, 3072 / 32), 256, 0, stream>>>(w_fcp, wt_fcp, 3072, 768);

    // 1) h = LN1(x)  (bf16)
    ln_kernel<<<dim3((unsigned)BS), 256, 0, stream>>>(x, ln1_w, ln1_b, h);
    // 2) qkv = h @ w_attn + b_attn  (bf16 out)
    gemm_kernel<false, false, true><<<dim3(2304 / BN, (unsigned)(BS / BM)), 256, 0, stream>>>(
        h, wt_attn, b_attn, nullptr, qkv, (int)BS, 2304, 768);
    // 3) vt = transpose(V)
    vtrans_kernel<<<dim3(S / 32, 2, Bb * 12), 256, 0, stream>>>(qkv, vt);
    // 4) attnout = causal_attention(qkv, vt)  (bf16)
    attn_kernel<<<dim3(S / 64, Bb * 12), 128, 0, stream>>>(qkv, vt, attnout);
    // 5) x1 = x + attnout @ w_proj + b_proj  (fp32)
    gemm_kernel<true, false, false><<<dim3(768 / BN, (unsigned)(BS / BM)), 256, 0, stream>>>(
        attnout, wt_proj, b_proj, x, x1, (int)BS, 768, 768);
    // 6) h = LN2(x1)  (bf16)
    ln_kernel<<<dim3((unsigned)BS), 256, 0, stream>>>(x1, ln2_w, ln2_b, h);
    // 7) fc = gelu(h @ w_fc + b_fc)  (bf16)
    gemm_kernel<false, true, true><<<dim3(3072 / BN, (unsigned)(BS / BM)), 256, 0, stream>>>(
        h, wt_fc, b_fc, nullptr, fc, (int)BS, 3072, 768);
    // 8) out = x1 + fc @ w_fc_proj + b_fc_proj  (fp32)
    gemm_kernel<true, false, false><<<dim3(768 / BN, (unsigned)(BS / BM)), 256, 0, stream>>>(
        fc, wt_fcp, b_fcp, x1, (float*)d_out, (int)BS, 768, 3072);
}
